// SparseMoELayer_44882408243215
// MI455X (gfx1250) — matrix operand order repaired and checked by reading, not compiled, not run on hardware
//
#include <hip/hip_runtime.h>
#include <hip/hip_bf16.h>

// SparseMoE top-1 switch routing for MI455X (gfx1250, wave32).
// bf16x3 split-precision WMMA (v_wmma_f32_16x16x32_bf16). B fragments are
// staged into LDS once per block (all 8 waves share them) with
// global_load_async_to_lds_b128 + s_wait_asynccnt double buffering.

#define NTOK  16384          // 4 * 4096 tokens
#define DD    1024
#define EE    8
#define KTILES 32            // 1024 / 32
#define NCHUNK 8             // n-chunks
#define NJ     8             // n-tiles per chunk
#define NITER  (NCHUNK * KTILES)   // 256 staged k-steps
#define STAGE_BYTES 16384    // 16 frags (8 n-tiles x hi/lo) * 1KB

typedef __attribute__((ext_vector_type(16))) __bf16 v16bf;
typedef __attribute__((ext_vector_type(8)))  float  v8f;

union FragB { uint4 q[2]; v16bf v; };
union FragA { unsigned u[8]; v16bf v; };

__device__ __forceinline__ unsigned rne_bf16(float f) {
  unsigned u = __float_as_uint(f);
  return (u + 0x7FFFu + ((u >> 16) & 1u)) >> 16;
}

// Split two fp32 into packed bf16 hi and lo dwords (elem a in [15:0]).
__device__ __forceinline__ void split2(float a, float b, unsigned& hi, unsigned& lo) {
  unsigned ha = rne_bf16(a), hb = rne_bf16(b);
  float ra = a - __uint_as_float(ha << 16);
  float rb = b - __uint_as_float(hb << 16);
  unsigned la = rne_bf16(ra), lb = rne_bf16(rb);
  hi = ha | (hb << 16);
  lo = la | (lb << 16);
}

// ---------------------------------------------------------------------------
// Prep: convert Wg [D,D] and We [E,D,D] fp32 -> bf16 hi/lo planes laid out as
// B-fragments: frag (n_tile, k_tile) = 32 lanes x 8 dwords, lane = khalf*16 +
// (n&15), dword v<4 holds K = khalf*8+2v,+1; v>=4 holds K = 16+khalf*8+2(v-4),+1.
// ---------------------------------------------------------------------------
__global__ __launch_bounds__(256) void moe_prep(
    const float* __restrict__ Wg, const float* __restrict__ We,
    unsigned* __restrict__ wgh, unsigned* __restrict__ wgl,
    unsigned* __restrict__ weh, unsigned* __restrict__ wel)
{
  const size_t PLANE = (size_t)DD * DD / 2;  // packed dwords per matrix
  size_t tid = (size_t)blockIdx.x * blockDim.x + threadIdx.x;
  if (tid >= (size_t)(1 + EE) * PLANE) return;
  int plane = (int)(tid / PLANE);
  int rem   = (int)(tid % PLANE);
  int kp = rem >> 10;          // k/2
  int n  = rem & (DD - 1);
  int k  = kp << 1;

  const float* src = (plane == 0) ? Wg : (We + (size_t)(plane - 1) * DD * DD);
  float a = src[(size_t)k * DD + n];
  float b = src[(size_t)(k + 1) * DD + n];
  unsigned hi, lo;
  split2(a, b, hi, lo);

  int ntile = n >> 4, ktile = k >> 5;
  int kr = k & 31, khalf = (kr >> 3) & 1, kk = (kr & 7) | ((kr >> 4) << 3), v = kk >> 1;
  int lanei = khalf * 16 + (n & 15);
  size_t dw = ((size_t)(ntile * KTILES + ktile) * 32 + lanei) * 8 + v;

  unsigned* dh = (plane == 0) ? wgh : (weh + (size_t)(plane - 1) * PLANE);
  unsigned* dl = (plane == 0) ? wgl : (wel + (size_t)(plane - 1) * PLANE);
  dh[dw] = hi;
  dl[dw] = lo;
}

// Build A hi/lo fragments from a row-major fp32 row (16-bit A 16x32 layout:
// lanes<16 hold K{0..7,16..23}; lanes>=16 hold K{8..15,24..31}).
__device__ __forceinline__ void load_a(const float* __restrict__ xr, int koff,
                                       FragA& ah, FragA& al) {
  float4 f0 = *(const float4*)(xr + koff);
  float4 f1 = *(const float4*)(xr + koff + 4);
  float4 f2 = *(const float4*)(xr + koff + 16);
  float4 f3 = *(const float4*)(xr + koff + 20);
  split2(f0.x, f0.y, ah.u[0], al.u[0]);
  split2(f0.z, f0.w, ah.u[1], al.u[1]);
  split2(f1.x, f1.y, ah.u[2], al.u[2]);
  split2(f1.z, f1.w, ah.u[3], al.u[3]);
  split2(f2.x, f2.y, ah.u[4], al.u[4]);
  split2(f2.z, f2.w, ah.u[5], al.u[5]);
  split2(f3.x, f3.y, ah.u[6], al.u[6]);
  split2(f3.z, f3.w, ah.u[7], al.u[7]);
}

// Stage the 16 KB of B fragments for one (nc, kt) step into an LDS buffer via
// async global->LDS copies: 1024 x 16B chunks across 256 threads.
// LDS layout: frag (j*2 + plane) at frag*1024 + lane*32 (+16 for 2nd quad).
__device__ __forceinline__ void issue_stage(
    const unsigned* __restrict__ wh, const unsigned* __restrict__ wl,
    unsigned char* buf, int nc, int kt, int tid)
{
#pragma unroll
  for (int i = 0; i < 4; ++i) {
    int c = tid + i * 256;
    int frag  = c >> 6;           // j*2 + plane
    int w     = c & 63;
    int lanei = w >> 1;
    int part  = (w & 1) << 2;     // dword offset within lane's 8 dwords
    int j     = frag >> 1;
    const unsigned* src = ((frag & 1) ? wl : wh)
        + ((size_t)((nc * NJ + j) * KTILES + kt) * 32 + lanei) * 8 + part;
    unsigned lds = (unsigned)(unsigned long long)(buf + c * 16);
    unsigned long long ga = (unsigned long long)src;
    asm volatile("global_load_async_to_lds_b128 %0, %1, off"
                 :: "v"(lds), "v"(ga) : "memory");
  }
}

__device__ __forceinline__ void wait_async_lds() {
  asm volatile("s_wait_asynccnt 0x0" ::: "memory");
}

// ---------------------------------------------------------------------------
// Gating: logits = x@Wg + bg, fused online softmax-max (m, sum, argmax) over
// the 1024 columns; p = 1/sum; route token to expert argmax % 8 via atomic
// compaction. One wave = 16 tokens; block = 8 waves = 128 tokens; all waves
// share LDS-staged Wg fragments.
// ---------------------------------------------------------------------------
__global__ __launch_bounds__(256) void moe_gating(
    const float* __restrict__ x, const unsigned* __restrict__ wgh,
    const unsigned* __restrict__ wgl, const float* __restrict__ bg,
    float* __restrict__ p_out, int* __restrict__ counts, int* __restrict__ lists)
{
  __shared__ __align__(16) unsigned char ldsB[2][STAGE_BYTES];

  const int tid  = threadIdx.x;
  const int lane = tid & 31;
  const int wave = tid >> 5;
  const int tok0 = blockIdx.x * 128 + wave * 16;
  const int half = lane >> 4;
  const int lc   = lane & 15;

  const float* xr = x + (size_t)(tok0 + lc) * DD;
  const int kbase = half * 8;

  float m[8], s[8];
  int   idx[8];
#pragma unroll
  for (int r = 0; r < 8; ++r) { m[r] = -3.0e38f; s[r] = 0.0f; idx[r] = 0; }

  const v8f vzero = {0.f, 0.f, 0.f, 0.f, 0.f, 0.f, 0.f, 0.f};
  v8f acc[NJ];
#pragma unroll
  for (int j = 0; j < NJ; ++j) acc[j] = vzero;

  issue_stage(wgh, wgl, ldsB[0], 0, 0, tid);

  for (int it = 0; it < NITER; ++it) {
    const int nc = it >> 5, kt = it & 31, cur = it & 1;
    wait_async_lds();
    __syncthreads();  // cur data visible to all; buf[1-cur] free (it-1 done)
    if (it + 1 < NITER)
      issue_stage(wgh, wgl, ldsB[cur ^ 1], (it + 1) >> 5, (it + 1) & 31, tid);

    FragA ah, al;
    load_a(xr, kt * 32 + kbase, ah, al);
#pragma unroll
    for (int j = 0; j < NJ; ++j) {
      const unsigned char* fp = ldsB[cur] + (size_t)j * 2048 + lane * 32;
      FragB bh = *(const FragB*)(fp);
      FragB bl = *(const FragB*)(fp + 1024);
      acc[j] = __builtin_amdgcn_wmma_f32_16x16x32_bf16(false, ah.v, false, bh.v, (short)0, acc[j], false, false);
      acc[j] = __builtin_amdgcn_wmma_f32_16x16x32_bf16(false, al.v, false, bh.v, (short)0, acc[j], false, false);
      acc[j] = __builtin_amdgcn_wmma_f32_16x16x32_bf16(false, ah.v, false, bl.v, (short)0, acc[j], false, false);
    }

    if (kt == 31) {  // online softmax update over this chunk's 8 n-tiles
#pragma unroll
      for (int j = 0; j < NJ; ++j) {
        const int n = (nc * NJ + j) * 16 + lc;
        const float bias = bg[n];
#pragma unroll
        for (int r = 0; r < 8; ++r) {
          float l = acc[j][r] + bias;
          if (l > m[r]) {
            s[r] = s[r] * __expf(m[r] - l) + 1.0f;
            m[r] = l;
            idx[r] = n;
          } else {
            s[r] += __expf(l - m[r]);
          }
        }
        acc[j] = vzero;
      }
    }
  }

  // Butterfly merge across the 16-lane half owning each row.
#pragma unroll
  for (int r = 0; r < 8; ++r) {
    for (int off = 1; off < 16; off <<= 1) {
      float m2 = __shfl_xor(m[r], off);
      float s2 = __shfl_xor(s[r], off);
      int   i2 = __shfl_xor(idx[r], off);
      float M  = fmaxf(m[r], m2);
      float sn = s[r] * __expf(m[r] - M) + s2 * __expf(m2 - M);
      int   iN = (m2 > m[r] || (m2 == m[r] && i2 < idx[r])) ? i2 : idx[r];
      m[r] = M; s[r] = sn; idx[r] = iN;
    }
  }

  if (lc == 0) {
#pragma unroll
    for (int r = 0; r < 8; ++r) {
      int token = tok0 + half * 8 + r;
      p_out[token] = 1.0f / s[r];       // exp(m-m)/sum = top-1 softmax prob
      int e = idx[r] & (EE - 1);
      int pos = atomicAdd(&counts[e], 1);
      lists[e * NTOK + pos] = token;
    }
  }
}

// ---------------------------------------------------------------------------
// Expert: block = (128 list positions) of one expert; 8 waves x 16 tokens.
// Gather rows from compacted list, out[t,:] = relu(x[t,:]@We[e]+be[e])*p[t].
// Block-uniform early exit so every wave participates in LDS barriers.
// ---------------------------------------------------------------------------
__global__ __launch_bounds__(256) void moe_expert(
    const float* __restrict__ x, const unsigned* __restrict__ weh,
    const unsigned* __restrict__ wel, const float* __restrict__ be,
    const float* __restrict__ p_in, const int* __restrict__ counts,
    const int* __restrict__ lists, float* __restrict__ out)
{
  __shared__ __align__(16) unsigned char ldsB[2][STAGE_BYTES];

  const int tid  = threadIdx.x;
  const int lane = tid & 31;
  const int wave = tid >> 5;
  const int e = blockIdx.y;
  const int cnt = counts[e];
  if (blockIdx.x * 128 >= cnt) return;   // block-uniform early exit

  const int start = blockIdx.x * 128 + wave * 16;
  const int half = lane >> 4;
  const int lc   = lane & 15;
  const int* lst = lists + e * NTOK;

  int ai = start + lc;
  if (ai >= cnt) ai = cnt - 1;                     // clamp pad rows
  const float* xr = x + (size_t)lst[ai] * DD;
  const int kbase = half * 8;

  int trow[8]; float pr[8]; unsigned vmask = 0;
#pragma unroll
  for (int r = 0; r < 8; ++r) {
    int Mi = start + half * 8 + r;
    if (Mi < cnt) { trow[r] = lst[Mi]; pr[r] = p_in[trow[r]]; vmask |= 1u << r; }
    else          { trow[r] = 0;       pr[r] = 0.0f; }
  }

  const size_t PLANE = (size_t)DD * DD / 2;
  const unsigned* wH = weh + (size_t)e * PLANE;
  const unsigned* wL = wel + (size_t)e * PLANE;
  const float* beE = be + e * DD;

  const v8f vzero = {0.f, 0.f, 0.f, 0.f, 0.f, 0.f, 0.f, 0.f};
  v8f acc[NJ];
#pragma unroll
  for (int j = 0; j < NJ; ++j) acc[j] = vzero;

  issue_stage(wH, wL, ldsB[0], 0, 0, tid);

  for (int it = 0; it < NITER; ++it) {
    const int nc = it >> 5, kt = it & 31, cur = it & 1;
    wait_async_lds();
    __syncthreads();
    if (it + 1 < NITER)
      issue_stage(wH, wL, ldsB[cur ^ 1], (it + 1) >> 5, (it + 1) & 31, tid);

    FragA ah, al;
    load_a(xr, kt * 32 + kbase, ah, al);
#pragma unroll
    for (int j = 0; j < NJ; ++j) {
      const unsigned char* fp = ldsB[cur] + (size_t)j * 2048 + lane * 32;
      FragB bh = *(const FragB*)(fp);
      FragB bl = *(const FragB*)(fp + 1024);
      acc[j] = __builtin_amdgcn_wmma_f32_16x16x32_bf16(false, ah.v, false, bh.v, (short)0, acc[j], false, false);
      acc[j] = __builtin_amdgcn_wmma_f32_16x16x32_bf16(false, al.v, false, bh.v, (short)0, acc[j], false, false);
      acc[j] = __builtin_amdgcn_wmma_f32_16x16x32_bf16(false, ah.v, false, bl.v, (short)0, acc[j], false, false);
    }

    if (kt == 31) {  // epilogue: bias + relu + gate-prob scale, masked scatter
#pragma unroll
      for (int j = 0; j < NJ; ++j) {
        const int n = (nc * NJ + j) * 16 + lc;
        const float bias = beE[n];
#pragma unroll
        for (int r = 0; r < 8; ++r) {
          if (vmask & (1u << r)) {
            float v = acc[j][r] + bias;
            v = v > 0.0f ? v : 0.0f;
            out[(size_t)trow[r] * DD + n] = v * pr[r];
          }
        }
        acc[j] = vzero;
      }
    }
  }
}

extern "C" void kernel_launch(void* const* d_in, const int* in_sizes, int n_in,
                              void* d_out, int out_size, void* d_ws, size_t ws_size,
                              hipStream_t stream) {
  const float* x  = (const float*)d_in[0];
  const float* Wg = (const float*)d_in[1];
  const float* bg = (const float*)d_in[2];
  const float* We = (const float*)d_in[3];
  const float* be = (const float*)d_in[4];
  float* out = (float*)d_out;

  const size_t PLANE = (size_t)DD * DD / 2;      // packed dwords per matrix
  unsigned* wgh = (unsigned*)d_ws;               // 2 MB
  unsigned* wgl = wgh + PLANE;                   // 2 MB
  unsigned* weh = wgl + PLANE;                   // 16 MB
  unsigned* wel = weh + EE * PLANE;              // 16 MB
  float* p_out  = (float*)(wel + EE * PLANE);    // 64 KB
  int* counts   = (int*)(p_out + NTOK);          // 8 ints (padded)
  int* lists    = counts + 64;                   // 512 KB

  hipMemsetAsync(counts, 0, 64 * sizeof(int), stream);

  size_t prep_threads = (size_t)(1 + EE) * PLANE;
  moe_prep<<<(unsigned)((prep_threads + 255) / 256), 256, 0, stream>>>(
      Wg, We, wgh, wgl, weh, wel);

  moe_gating<<<NTOK / 128, 256, 0, stream>>>(x, wgh, wgl, bg, p_out, counts, lists);

  dim3 grid(NTOK / 128, EE);
  moe_expert<<<grid, 256, 0, stream>>>(x, weh, wel, be, p_out, counts, lists, out);
}
